// VoxelBackBone8xTSSPV_72688026518170
// MI455X (gfx1250) — compile-verified, matrix operand order checked by reading, not run on hardware
//
#include <hip/hip_runtime.h>
#include <hip/hip_bf16.h>

// ---------------------------------------------------------------------------
// MI455X (gfx1250) VoxelBackBone8x as implicit-GEMM WMMA, v3.
//
//  * Activations f16 channel-last with 1-voxel zero halo (unconditional taps).
//  * ALL geometry is template-constant -> every tap/B/mask load folds its
//    offset into the global-load 24-bit immediate (no per-tap VALU).
//  * Unified tap-major GEMM-K: Kg = tap*CIN + ci, padded to 32. Dense K for
//    CIN=4/16 (no zero-padded WMMA lanes): conv_input 27->4 WMMAs/tile,
//    conv1 & conv2[0] 27->14 chunks. CIN>=32 unchanged (1 chunk = 32 of Cin).
//  * A fragments in CDNA5 16-bit A layout, staged in LDS (<56KB layers).
//  * v_wmma_f32_16x16x32_f16, f32 accum; BN+ReLU+mask fused epilogue.
// ---------------------------------------------------------------------------

typedef __attribute__((ext_vector_type(16))) _Float16 v16h;
typedef __attribute__((ext_vector_type(8)))  _Float16 v8h;
typedef __attribute__((ext_vector_type(4)))  _Float16 v4h;
typedef __attribute__((ext_vector_type(8)))  float    v8f;

__device__ __forceinline__ v8f wmma_f16(v16h a, v16h b, v8f c) {
  return __builtin_amdgcn_wmma_f32_16x16x32_f16(false, a, false, b, (short)0, c,
                                                false, false);
}

// ---------------- setup kernels ----------------

__global__ void fill_u128_kernel(uint4* __restrict__ p, unsigned int val, long n16) {
  long i = (long)blockIdx.x * blockDim.x + threadIdx.x;
  if (i >= n16) return;
  p[i] = make_uint4(val, val, val, val);
}

__global__ void scatter_win_kernel(const int* __restrict__ c, int* __restrict__ win,
                                   int n, int HP, int WP) {
  int i = blockIdx.x * blockDim.x + threadIdx.x;
  if (i >= n) return;
  int z = c[i * 4 + 1], y = c[i * 4 + 2], x = c[i * 4 + 3];
  int vox = ((z + 1) * HP + (y + 1)) * WP + (x + 1);
  atomicMax(&win[vox], i);  // deterministic "last index wins"
}

__global__ void scatter_feat_kernel(const float* __restrict__ f, const int* __restrict__ c,
                                    const int* __restrict__ win, _Float16* __restrict__ g,
                                    unsigned char* __restrict__ m, int n, int HP, int WP) {
  int i = blockIdx.x * blockDim.x + threadIdx.x;
  if (i >= n) return;
  int z = c[i * 4 + 1], y = c[i * 4 + 2], x = c[i * 4 + 3];
  int vox = ((z + 1) * HP + (y + 1)) * WP + (x + 1);
  if (win[vox] != i) return;
  v4h t;
  t[0] = (_Float16)f[i * 4 + 0];
  t[1] = (_Float16)f[i * 4 + 1];
  t[2] = (_Float16)f[i * 4 + 2];
  t[3] = (_Float16)f[i * 4 + 3];
  *(v4h*)(g + (long)vox * 4) = t;
  m[vox] = 1;
}

__global__ void bn_prep_kernel(const float* __restrict__ g, const float* __restrict__ b,
                               const float* __restrict__ mn, const float* __restrict__ vr,
                               float* __restrict__ sb, int C) {
  int i = blockIdx.x * blockDim.x + threadIdx.x;
  if (i >= C) return;
  float s = g[i] * rsqrtf(vr[i] + 1e-3f);
  sb[i] = s;
  sb[C + i] = b[i] - mn[i] * s;
}

// Repack OIDHW f32 weights into WMMA A fragments (f16), unified tap-major K:
// Kg = ch*32 + K(lane,h); tap = Kg / I; ci = Kg % I; frag = ch*OC + oc.
__global__ void pack_w_kernel(const float* __restrict__ w, _Float16* __restrict__ dst,
                              int O, int I, int T) {
  int OC = O >> 4;
  int NCH = (T * I + 31) >> 5;
  long total = (long)NCH * OC * 512;
  long idx = (long)blockIdx.x * blockDim.x + threadIdx.x;
  if (idx >= total) return;
  int h    = (int)(idx & 15);
  int lane = (int)((idx >> 4) & 31);
  long frag = idx >> 9;
  int oc = (int)(frag % OC);
  int ch = (int)(frag / OC);
  int M  = oc * 16 + (lane & 15);
  int K  = (lane < 16) ? ((h < 8) ? h : h + 8) : ((h < 8) ? h + 8 : h + 16);
  int Kg = ch * 32 + K;
  int t  = Kg / I;
  int ci = Kg % I;
  float val = (t < T) ? w[((long)M * I + ci) * T + t] : 0.0f;
  dst[idx] = (_Float16)val;
}

// ---------------- the conv kernel ----------------
// One wave32 per tile of 16 consecutive output-x at fixed (z_out, y_out).

template <int CIN, int COUT, int KD, int KH, int KW, int SD, int SH, int SW,
          int PD, int PH, int PW, int DZ, int HY, int WX, int DO, int HO, int WO,
          bool DOWN, bool FINAL, bool USE_LDS>
__global__ __launch_bounds__(256) void conv3d_wmma(
    const _Float16* __restrict__ in,          // padded (DZ+2)(HY+2)(WX+2) x CIN
    const unsigned char* __restrict__ inMask, // padded
    const _Float16* __restrict__ wpack, const float* __restrict__ bnsb,
    _Float16* __restrict__ out,               // padded (DO+2)(HO+2)(WO+2) x COUT
    unsigned char* __restrict__ outMask,      // padded
    float* __restrict__ outF32) {             // unpadded NCDHW (FINAL only)
  constexpr int OC    = COUT / 16;
  constexpr int T     = KD * KH * KW;
  constexpr int NCH   = (T * CIN + 31) / 32;  // GEMM-K chunks of 32
  constexpr int NFRAG = NCH * OC;
  constexpr int HP = HY + 2, WP = WX + 2, HOP = HO + 2, WOP = WO + 2;
  constexpr int WT = (WO + 15) / 16;
  constexpr int TOT = DO * HO * WT;

  __shared__ __align__(32) _Float16 smem[USE_LDS ? NFRAG * 512 : 8];
  if constexpr (USE_LDS) {
    for (int i = threadIdx.x; i < NFRAG * 64; i += 256)
      *(v8h*)(smem + i * 8) = *(const v8h*)(wpack + i * 8);
    __syncthreads();
  }

  const int lane = threadIdx.x & 31;
  const int col  = lane & 15;
  const int hi   = lane >> 4;
  const int tile =
      __builtin_amdgcn_readfirstlane((int)blockIdx.x * 8 + (int)(threadIdx.x >> 5));
  if (tile >= TOT) return;  // whole-wave exit; no barriers after this

  const int wt = tile % WT;
  const int zy = tile / WT;
  const int yo = zy % HO;
  const int zo = zy / HO;
  const int xo = wt * 16 + col;
  const bool colValid = (xo < WO);
  const int xoc = colValid ? xo : (WO - 1);   // clamp -> stays inside halo
  const int tileVox = zo * SD * HP * WP + yo * SH * WP;  // uniform (SGPR)
  const unsigned int hiMask = hi ? 0u : 0xFFFFFFFFu;

  // per-lane B base; all tap offsets are compile-time immediates from here
  const _Float16* bptr =
      in + tileVox * CIN + xoc * SW * CIN + (CIN >= 16 ? hi * 8 : 0);

  v8f acc[OC] = {};

#pragma unroll
  for (int ch = 0; ch < NCH; ++ch) {
    v16h b;
    if constexpr (CIN == 4) {
      // chunk = 8 taps; lane half-groups of 4 = one tap's 4 channels
#pragma unroll
      for (int s = 0; s < 2; ++s) {
#pragma unroll
        for (int q = 0; q < 2; ++q) {
          const int tl = ch * 8 + s * 4 + q;  // tap for hi==0 lanes
          const int th = tl + 2;              // tap for hi==1 lanes
          uint2 u = make_uint2(0u, 0u);
          if (tl < T && th < T) {
            const int oL = ((((tl / (KH * KW)) - PD + 1) * HP +
                             ((tl / KW) % KH) - PH + 1) * WP + (tl % KW) - PW + 1) * CIN;
            const int oH = ((((th / (KH * KW)) - PD + 1) * HP +
                             ((th / KW) % KH) - PH + 1) * WP + (th % KW) - PW + 1) * CIN;
            u = *(const uint2*)(bptr + (hi ? oH : oL));
          } else if (tl < T) {
            const int oL = ((((tl / (KH * KW)) - PD + 1) * HP +
                             ((tl / KW) % KH) - PH + 1) * WP + (tl % KW) - PW + 1) * CIN;
            u = *(const uint2*)(bptr + oL);
            u.x &= hiMask;  // hi lanes' tap is out of range -> exact zeros
            u.y &= hiMask;
          }
          v4h t4 = __builtin_bit_cast(v4h, u);
          b[s * 8 + q * 4 + 0] = t4[0];
          b[s * 8 + q * 4 + 1] = t4[1];
          b[s * 8 + q * 4 + 2] = t4[2];
          b[s * 8 + q * 4 + 3] = t4[3];
        }
      }
    } else if constexpr (CIN == 16) {
      // chunk = 2 taps; each lane segment is one v8h at a constant offset
#pragma unroll
      for (int s = 0; s < 2; ++s) {
        const int t = ch * 2 + s;
        v8h seg = {};
        if (t < T) {
          const int kd = t / (KH * KW), kh = (t / KW) % KH, kw = t % KW;
          const int off =
              (((kd - PD + 1) * HP + (kh - PH + 1)) * WP + (kw - PW + 1)) * CIN;
          seg = *(const v8h*)(bptr + off);
        }
#pragma unroll
        for (int j = 0; j < 8; ++j) b[s * 8 + j] = seg[j];
      }
    } else {
      // CIN multiple of 32: chunk = 32 channels of one tap
      constexpr int KCC = CIN / 32;
      const int t = ch / KCC, kc = ch % KCC;
      const int kd = t / (KH * KW), kh = (t / KW) % KH, kw = t % KW;
      const int off =
          (((kd - PD + 1) * HP + (kh - PH + 1)) * WP + (kw - PW + 1)) * CIN + kc * 32;
      v8h lo = *(const v8h*)(bptr + off);
      v8h hh = *(const v8h*)(bptr + off + 16);
#pragma unroll
      for (int j = 0; j < 8; ++j) { b[j] = lo[j]; b[8 + j] = hh[j]; }
    }
#pragma unroll
    for (int o = 0; o < OC; ++o) {
      const int fofs = (ch * OC + o) * 512 + lane * 16;
      v16h a;
      if constexpr (USE_LDS) a = *(const v16h*)(smem + fofs);
      else                   a = *(const v16h*)(wpack + fofs);
      acc[o] = wmma_f16(a, b, acc[o]);
    }
  }

  // ---- mask ----
  bool act;
  if constexpr (DOWN) {
    const unsigned char* mp = inMask + tileVox + xoc * SW;
    bool maskAny = false;
#pragma unroll
    for (int t = 0; t < T; ++t) {
      const int kd = t / (KH * KW), kh = (t / KW) % KH, kw = t % KW;
      const int off = ((kd - PD + 1) * HP + (kh - PH + 1)) * WP + (kw - PW + 1);
      maskAny |= (mp[off] != 0);
    }
    act = maskAny;
    const int ovoxM = ((zo + 1) * HOP + (yo + 1)) * WOP + (xoc + 1);
    if (outMask != nullptr && hi == 0 && colValid) outMask[ovoxM] = act ? 1 : 0;
  } else {
    const int ovoxIn = ((zo + 1) * HP + (yo + 1)) * WP + (xoc + 1);
    act = colValid && (inMask[ovoxIn] != 0);
  }

  // ---- BN + ReLU + store ----
#pragma unroll
  for (int o = 0; o < OC; ++o) {
    const v8f sc = *(const v8f*)(bnsb + o * 16 + hi * 8);
    const v8f bi = *(const v8f*)(bnsb + COUT + o * 16 + hi * 8);
    if constexpr (FINAL) {
      if (colValid) {
        constexpr int plane = DO * HO * WO;
        const int lvox = (zo * HO + yo) * WO + xo;  // unpadded NCDHW
#pragma unroll
        for (int r = 0; r < 8; ++r) {
          float v = acc[o][r] * sc[r] + bi[r];
          v = act ? fmaxf(v, 0.0f) : 0.0f;
          outF32[(o * 16 + hi * 8 + r) * plane + lvox] = v;
        }
      }
    } else {
      v8h hv;
#pragma unroll
      for (int r = 0; r < 8; ++r) {
        float v = acc[o][r] * sc[r] + bi[r];
        v = act ? fmaxf(v, 0.0f) : 0.0f;
        hv[r] = (_Float16)v;
      }
      const int ovox = ((zo + 1) * HOP + (yo + 1)) * WOP + (xoc + 1);
      if (colValid) *(v8h*)(out + (long)ovox * COUT + o * 16 + hi * 8) = hv;
    }
  }
}

// ---------------- host-side orchestration ----------------

extern "C" void kernel_launch(void* const* d_in, const int* in_sizes, int n_in,
                              void* d_out, int out_size, void* d_ws, size_t ws_size,
                              hipStream_t stream) {
  (void)n_in; (void)out_size; (void)ws_size;
  const float* feat = (const float*)d_in[0];
  const int* coords = (const int*)d_in[1];
  const int N = in_sizes[0] / 4;

  const int D0 = 41, H0 = 160, W0 = 160;
  const int D2 = 21, H2 = 80,  W2 = 80;
  const int D3 = 11, H3 = 40,  W3 = 40;
  const int D4 = 5,  H4 = 20,  W4 = 20;
  const long P0 = (long)(D0 + 2) * (H0 + 2) * (W0 + 2);
  const long P2 = (long)(D2 + 2) * (H2 + 2) * (W2 + 2);
  const long P3 = (long)(D3 + 2) * (H3 + 2) * (W3 + 2);
  const long P4 = (long)(D4 + 2) * (H4 + 2) * (W4 + 2);

  // d_in flattening: insertion order; per block {w, gamma, beta, mean, var}.
  struct LInfo { int w, g, b, m, v, O, I, T; };
  const LInfo L[12] = {
      { 2,  3,  4,  5,  6,  16,  4, 27},   // conv_input
      { 7,  8,  9, 10, 11,  16, 16, 27},   // conv1[0]
      {12, 13, 14, 15, 16,  32, 16, 27},   // conv2[0] down
      {17, 18, 19, 20, 21,  32, 32, 27},   // conv2[1]
      {22, 23, 24, 25, 26,  32, 32, 27},   // conv2[2]
      {27, 28, 29, 30, 31,  64, 32, 27},   // conv3[0] down
      {32, 33, 34, 35, 36,  64, 64, 27},   // conv3[1]
      {37, 38, 39, 40, 41,  64, 64, 27},   // conv3[2]
      {42, 43, 44, 45, 46,  64, 64, 27},   // conv4[0] down
      {47, 48, 49, 50, 51,  64, 64, 27},   // conv4[1]
      {52, 53, 54, 55, 56,  64, 64, 27},   // conv4[2]
      {57, 58, 59, 60, 61, 128, 64,  3},   // conv_out
  };

  char* ws = (char*)d_ws;
  size_t off = 0;
  auto alloc = [&](size_t bytes) -> void* {
    void* p = ws + off;
    off = (off + bytes + 255) & ~(size_t)255;
    return p;
  };
  auto fill = [&](void* p, size_t bytes, unsigned int val) {
    long n16 = (long)((bytes + 15) / 16);
    fill_u128_kernel<<<(unsigned)((n16 + 255) / 256), 256, 0, stream>>>((uint4*)p, val, n16);
  };

  _Float16* grid0 = (_Float16*)alloc(P0 * 4 * 2);
  unsigned char* m0 = (unsigned char*)alloc(P0);
  int* win0 = (int*)alloc(P0 * 4);
  _Float16* bA  = (_Float16*)alloc(P0 * 16 * 2);
  _Float16* bB  = (_Float16*)alloc(P0 * 16 * 2);
  _Float16* c2A = (_Float16*)alloc(P2 * 32 * 2);
  _Float16* c2B = (_Float16*)alloc(P2 * 32 * 2);
  unsigned char* m2 = (unsigned char*)alloc(P2);
  _Float16* c3A = (_Float16*)alloc(P3 * 64 * 2);
  _Float16* c3B = (_Float16*)alloc(P3 * 64 * 2);
  unsigned char* m3 = (unsigned char*)alloc(P3);
  _Float16* c4A = (_Float16*)alloc(P4 * 64 * 2);
  _Float16* c4B = (_Float16*)alloc(P4 * 64 * 2);
  unsigned char* m4 = (unsigned char*)alloc(P4);

  _Float16* pw[12];
  float* sb[12];
  for (int i = 0; i < 12; ++i) {
    int NCH = (L[i].T * L[i].I + 31) >> 5;
    long frags = (long)NCH * (L[i].O / 16);
    pw[i] = (_Float16*)alloc(frags * 512 * 2);
    sb[i] = (float*)alloc((size_t)L[i].O * 2 * 4);
  }

  // ---- zero halos/interiors, init winner buffer ----
  fill(grid0, P0 * 8, 0u);
  fill(m0, P0, 0u);
  fill(win0, P0 * 4, 0xFFFFFFFFu);
  fill(bA, P0 * 32, 0u);
  fill(bB, P0 * 32, 0u);
  fill(c2A, P2 * 64, 0u);  fill(c2B, P2 * 64, 0u);  fill(m2, P2, 0u);
  fill(c3A, P3 * 128, 0u); fill(c3B, P3 * 128, 0u); fill(m3, P3, 0u);
  fill(c4A, P4 * 128, 0u); fill(c4B, P4 * 128, 0u); fill(m4, P4, 0u);

  // ---- scatter (deterministic winner pass) ----
  scatter_win_kernel<<<(N + 255) / 256, 256, 0, stream>>>(coords, win0, N, H0 + 2, W0 + 2);
  scatter_feat_kernel<<<(N + 255) / 256, 256, 0, stream>>>(feat, coords, win0, grid0, m0,
                                                           N, H0 + 2, W0 + 2);

  // ---- per-layer param prep ----
  for (int i = 0; i < 12; ++i) {
    bn_prep_kernel<<<1, 128, 0, stream>>>((const float*)d_in[L[i].g], (const float*)d_in[L[i].b],
                                          (const float*)d_in[L[i].m], (const float*)d_in[L[i].v],
                                          sb[i], L[i].O);
    int NCH = (L[i].T * L[i].I + 31) >> 5;
    long total = (long)NCH * (L[i].O / 16) * 512;
    pack_w_kernel<<<(unsigned)((total + 255) / 256), 256, 0, stream>>>(
        (const float*)d_in[L[i].w], pw[i], L[i].O, L[i].I, L[i].T);
  }

  auto tiles = [](int Do, int Ho, int Wo) { return (long)Do * Ho * ((Wo + 15) / 16); };
  auto nb = [](long t) { return dim3((unsigned)((t + 7) / 8)); };  // 8 waves / block

  // ---- network ----
  long t0 = tiles(D0, H0, W0);
  conv3d_wmma<4, 16, 3, 3, 3, 1, 1, 1, 1, 1, 1, 41, 160, 160, 41, 160, 160,
              false, false, true><<<nb(t0), 256, 0, stream>>>(
      grid0, m0, pw[0], sb[0], bA, nullptr, nullptr);
  conv3d_wmma<16, 16, 3, 3, 3, 1, 1, 1, 1, 1, 1, 41, 160, 160, 41, 160, 160,
              false, false, true><<<nb(t0), 256, 0, stream>>>(
      bA, m0, pw[1], sb[1], bB, nullptr, nullptr);

  long t2 = tiles(D2, H2, W2);
  conv3d_wmma<16, 32, 3, 3, 3, 2, 2, 2, 1, 1, 1, 41, 160, 160, 21, 80, 80,
              true, false, true><<<nb(t2), 256, 0, stream>>>(
      bB, m0, pw[2], sb[2], c2A, m2, nullptr);
  conv3d_wmma<32, 32, 3, 3, 3, 1, 1, 1, 1, 1, 1, 21, 80, 80, 21, 80, 80,
              false, false, true><<<nb(t2), 256, 0, stream>>>(
      c2A, m2, pw[3], sb[3], c2B, nullptr, nullptr);
  conv3d_wmma<32, 32, 3, 3, 3, 1, 1, 1, 1, 1, 1, 21, 80, 80, 21, 80, 80,
              false, false, true><<<nb(t2), 256, 0, stream>>>(
      c2B, m2, pw[4], sb[4], c2A, nullptr, nullptr);

  long t3 = tiles(D3, H3, W3);
  conv3d_wmma<32, 64, 3, 3, 3, 2, 2, 2, 1, 1, 1, 21, 80, 80, 11, 40, 40,
              true, false, false><<<nb(t3), 256, 0, stream>>>(
      c2A, m2, pw[5], sb[5], c3A, m3, nullptr);
  conv3d_wmma<64, 64, 3, 3, 3, 1, 1, 1, 1, 1, 1, 11, 40, 40, 11, 40, 40,
              false, false, false><<<nb(t3), 256, 0, stream>>>(
      c3A, m3, pw[6], sb[6], c3B, nullptr, nullptr);
  conv3d_wmma<64, 64, 3, 3, 3, 1, 1, 1, 1, 1, 1, 11, 40, 40, 11, 40, 40,
              false, false, false><<<nb(t3), 256, 0, stream>>>(
      c3B, m3, pw[7], sb[7], c3A, nullptr, nullptr);

  long t4 = tiles(D4, H4, W4);
  conv3d_wmma<64, 64, 3, 3, 3, 2, 2, 2, 0, 1, 1, 11, 40, 40, 5, 20, 20,
              true, false, false><<<nb(t4), 256, 0, stream>>>(
      c3A, m3, pw[8], sb[8], c4A, m4, nullptr);
  conv3d_wmma<64, 64, 3, 3, 3, 1, 1, 1, 1, 1, 1, 5, 20, 20, 5, 20, 20,
              false, false, false><<<nb(t4), 256, 0, stream>>>(
      c4A, m4, pw[9], sb[9], c4B, nullptr, nullptr);
  conv3d_wmma<64, 64, 3, 3, 3, 1, 1, 1, 1, 1, 1, 5, 20, 20, 5, 20, 20,
              false, false, false><<<nb(t4), 256, 0, stream>>>(
      c4B, m4, pw[10], sb[10], c4A, nullptr, nullptr);

  long t5 = tiles(2, 20, 20);
  conv3d_wmma<64, 128, 3, 1, 1, 2, 1, 1, 0, 0, 0, 5, 20, 20, 2, 20, 20,
              true, true, true><<<nb(t5), 256, 0, stream>>>(
      c4A, m4, pw[11], sb[11], nullptr, nullptr, (float*)d_out);
}
